// ContextNetWeightModule_51891794870773
// MI455X (gfx1250) — matrix operand order repaired and checked by reading, not compiled, not run on hardware
//
#include <hip/hip_runtime.h>
#include <hip/hip_bf16.h>
#include <math.h>

typedef __attribute__((ext_vector_type(16))) _Float16 v16h;
typedef __attribute__((ext_vector_type(8)))  _Float16 v8h;
typedef __attribute__((ext_vector_type(8)))  float    v8f;

#define B_   32
#define C_   128
#define T_   8192
#define K_   5
#define MC_  256          // M*C  (fc1 outputs)
#define QD_  640          // GEMM inner dim = C*K, tap-major: q = k*128 + ci

// ---------------------------------------------------------------------------
// Kernel 1: masked global average pool.  One block per (b,c) row.
// ---------------------------------------------------------------------------
__global__ __launch_bounds__(256) void gap_kernel(const float* __restrict__ x,
                                                  const int* __restrict__ lens,
                                                  float* __restrict__ gap) {
    __shared__ float red[256];
    const int bc  = blockIdx.x;                 // 0 .. B*C-1
    const int tid = threadIdx.x;
    const float4* row = (const float4*)(x + (size_t)bc * T_);
    float s = 0.0f;
    for (int i = tid; i < T_ / 4; i += 256) {
        float4 v = row[i];
        s += v.x + v.y + v.z + v.w;
    }
    red[tid] = s;
    __syncthreads();
    for (int off = 128; off > 0; off >>= 1) {
        if (tid < off) red[tid] += red[tid + off];
        __syncthreads();
    }
    if (tid == 0) gap[bc] = red[0] / (float)lens[bc / C_];
}

// ---------------------------------------------------------------------------
// Kernel 2: h = sigmoid(gap @ w1^T + b1).  One block per sample.
// ---------------------------------------------------------------------------
__global__ __launch_bounds__(256) void fc1_kernel(const float* __restrict__ gap,
                                                  const float* __restrict__ w1,
                                                  const float* __restrict__ b1,
                                                  float* __restrict__ h) {
    __shared__ float sg[C_];
    const int b = blockIdx.x;
    const int j = threadIdx.x;                  // 0 .. MC_-1
    if (j < C_) sg[j] = gap[b * C_ + j];
    __syncthreads();
    float acc = b1[j];
    const float* wr = w1 + (size_t)j * C_;
    #pragma unroll 8
    for (int c = 0; c < C_; ++c) acc += sg[c] * wr[c];
    h[b * MC_ + j] = 1.0f / (1.0f + expf(-acc));
}

// ---------------------------------------------------------------------------
// Kernel 3: build per-sample dynamic weights in f16, tap-major layout:
//   a_dyn[b][co][q],  q = k*128 + ci,
//   value = h[b, 2*co + ((ci*5+k) >= 320)] * w2[co*640 + ci*5 + k]
// ---------------------------------------------------------------------------
__global__ __launch_bounds__(256) void wdyn_kernel(const float* __restrict__ h,
                                                   const float* __restrict__ w2,
                                                   _Float16* __restrict__ a_dyn) {
    const int idx = blockIdx.x * 256 + threadIdx.x;   // < B_*C_*QD_ = 2621440
    if (idx >= B_ * C_ * QD_) return;
    const int b  = idx / (C_ * QD_);
    const int r  = idx - b * (C_ * QD_);
    const int co = r / QD_;
    const int qp = r - co * QD_;
    const int k  = qp >> 7;                           // tap
    const int ci = qp & 127;
    const int f  = co * QD_ + ci * K_ + k;            // flat index per sample
    const float v = h[b * MC_ + (f / 320)] * w2[f];
    a_dyn[idx] = (_Float16)v;
}

// ---------------------------------------------------------------------------
// Kernel 4: out(128 x 128T-tile) = A(128 x 640) x B_im2col(640 x 128T-tile)
// per sample with v_wmma_f32_16x16x32_f16.  A chunks are double-buffered via
// GLOBAL_LOAD_ASYNC_TO_LDS_B128 (ASYNCcnt).  Grid = (B, T/128), 8 waves/WG.
// ---------------------------------------------------------------------------
__global__ __launch_bounds__(256) void conv_wmma_kernel(
        const float*    __restrict__ x,
        const _Float16* __restrict__ a_dyn,
        float*          __restrict__ out) {
    // x tile, transposed + f16: lds_xT[t_local][ci], t_local = 0..131 (halo 2+2)
    __shared__ __align__(16) _Float16 lds_xT[132][136];          // 35904 B
    // A chunks: 32 K-values per chunk, double buffered, row-major [co][q_local]
    __shared__ __align__(16) _Float16 lds_a[2][128][40];         // 2 x 10240 B

    const int b    = blockIdx.x;
    const int t0   = blockIdx.y * 128;
    const int tid  = threadIdx.x;
    const int lane = tid & 31;
    const int wave = tid >> 5;

    // ---- async A staging: 128 rows x 32 halves (8 KB) per chunk,
    //      each thread owns 32 B (two b128 async ops, shared address setup)
    const int tco  = tid >> 1;
    const int toff = (tid & 1) * 16;                  // halves
    const _Float16* gA = a_dyn + ((size_t)(b * C_ + tco)) * QD_ + toff;

    auto issue_chunk = [&](int c, int buf) {
        unsigned ldsb = (unsigned)(uintptr_t)&lds_a[buf][tco][toff];
        unsigned long long ga = (unsigned long long)(gA + c * 32);
        asm volatile(
            "global_load_async_to_lds_b128 %0, %1, off\n\t"
            "global_load_async_to_lds_b128 %0, %1, off offset:16"
            :: "v"(ldsb), "v"(ga) : "memory");
    };

    issue_chunk(0, 0);          // prefetch first A chunk while x tile stages

    // ---- stage x tile (f32 global -> f16 LDS, transposed), zero-pad edges
    for (int i = tid; i < C_ * 132; i += 256) {
        const int ci = i / 132;
        const int tl = i - ci * 132;
        const int t  = t0 - 2 + tl;
        float v = (t >= 0 && t < T_) ? x[((size_t)(b * C_ + ci)) * T_ + t] : 0.0f;
        lds_xT[tl][ci] = (_Float16)v;
    }

    v8f acc[8] = {};

    const int m0   = wave * 16;           // M strip of this wave
    const int arow = m0 + (lane & 15);    // A row this lane supplies
    const int qsel = (lane >> 4) * 16;    // A: lanes>=16 hold K 16..31
    const int ncol = lane & 15;           // B/D: column within 16-wide N tile
    const int csel = (lane >> 4) * 16;    // B: lanes>=16 hold K rows 16..31
    const int msel = (lane >> 4) * 8;     // D: lanes>=16 hold M rows 8..15

    for (int c = 0; c < QD_ / 32; ++c) {  // 20 chunks, 1 WMMA K-step each
        if (c < QD_ / 32 - 1) {
            issue_chunk(c + 1, (c + 1) & 1);          // overlap next chunk
            asm volatile("s_wait_asynccnt 2" ::: "memory");  // chunk c landed
        } else {
            asm volatile("s_wait_asynccnt 0" ::: "memory");
        }
        __syncthreads();                  // async LDS writes visible to all

        const int k_tap   = c >> 2;       // tap constant inside a chunk
        const int ci_base = (c & 3) * 32 + csel;

        // A fragment (reused across all 8 N tiles)
        const _Float16* ap = &lds_a[c & 1][arow][qsel];
        v8h a_lo = *(const v8h*)ap;            // K {0..7} or {16..23}
        v8h a_hi = *(const v8h*)(ap + 8);      // K {8..15} or {24..31}
        v16h afrag = __builtin_shufflevector(a_lo, a_hi,
            0,1,2,3,4,5,6,7,8,9,10,11,12,13,14,15);

        #pragma unroll
        for (int nbt = 0; nbt < 8; ++nbt) {
            const int tl = nbt * 16 + ncol + k_tap;
            const _Float16* bp = &lds_xT[tl][ci_base];
            v8h b_lo = *(const v8h*)bp;        // 16 contiguous ci (tap-major!)
            v8h b_hi = *(const v8h*)(bp + 8);
            v16h bfrag = __builtin_shufflevector(b_lo, b_hi,
                0,1,2,3,4,5,6,7,8,9,10,11,12,13,14,15);
            acc[nbt] = __builtin_amdgcn_wmma_f32_16x16x32_f16(
                false, afrag, false, bfrag, (short)0, acc[nbt], false, false);
        }

        __syncthreads();                  // all reads done before overwrite
    }

    // ---- write D tiles: VGPR r <-> M = m0 + msel + r ; column = t0+16*nbt+ncol
    #pragma unroll
    for (int nbt = 0; nbt < 8; ++nbt) {
        const int t = t0 + nbt * 16 + ncol;
        #pragma unroll
        for (int r = 0; r < 8; ++r) {
            const int row = m0 + msel + r;
            out[((size_t)(b * C_ + row)) * T_ + t] = acc[nbt][r];
        }
    }
}

// ---------------------------------------------------------------------------
extern "C" void kernel_launch(void* const* d_in, const int* in_sizes, int n_in,
                              void* d_out, int out_size, void* d_ws, size_t ws_size,
                              hipStream_t stream) {
    const float* x    = (const float*)d_in[0];
    const int*   lens = (const int*)d_in[1];
    const float* w1   = (const float*)d_in[2];
    const float* b1   = (const float*)d_in[3];
    const float* w2   = (const float*)d_in[4];
    float*       out  = (float*)d_out;

    // workspace layout
    float*    gapW = (float*)d_ws;                                    // 4096 f32
    float*    hW   = (float*)((char*)d_ws + 16384);                   // 8192 f32
    _Float16* aW   = (_Float16*)((char*)d_ws + 16384 + 32768);        // 2.62M f16

    gap_kernel <<<B_ * C_,              256, 0, stream>>>(x, lens, gapW);
    fc1_kernel <<<B_,                   256, 0, stream>>>(gapW, w1, b1, hW);
    wdyn_kernel<<<(B_*C_*QD_)/256,      256, 0, stream>>>(hW, w2, aW);
    dim3 grid(B_, T_ / 128);
    conv_wmma_kernel<<<grid, 256, 0, stream>>>(x, aW, out);
}
